// Chunker_32581621908230
// MI455X (gfx1250) — matrix operand order repaired and checked by reading, not compiled, not run on hardware
//
#include <hip/hip_runtime.h>

// ----------------------------------------------------------------------------
// Quantized linear: y = clip(round(x/scale)+zp) @ W^T + bias
//   x: (8192, 4096) f32   W: (4096, 4096) f32   bias: (4096) f32
// MI455X strategy:
//   1) global min/max of x (ordered-uint atomics)
//   2) quantize x -> f16 codes (integers in [-128,127] are exact in f16)
//   3) split W into f16 hi/lo:  W ~= Whi + Wlo  (double-f16 ~ 22 bits of W)
//   4) WMMA GEMM (v_wmma_f32_16x16x32_f16), LDS double-buffered via
//      GLOBAL_LOAD_ASYNC_TO_LDS_B128 (ASYNCcnt), one barrier per K-step.
// qx(64MiB)+Whi+Wlo(64MiB) fit in the 192MB L2 -> GEMM is compute-bound on
// the WMMA pipes; HBM traffic is ~450MiB (~19us at 23.3TB/s).
// Workspace layout: [0,8) min/max, +256: qx f16, then Whi f16, then Wlo f16.
// ----------------------------------------------------------------------------

typedef _Float16 v4h __attribute__((ext_vector_type(4)));
typedef _Float16 v8h __attribute__((ext_vector_type(8)));
typedef _Float16 v16h __attribute__((ext_vector_type(16)));
typedef float    v8f __attribute__((ext_vector_type(8)));

// ---- float <-> order-preserving uint (for atomic min/max on f32) -----------
__device__ __forceinline__ unsigned f2o(float f) {
  unsigned u = __float_as_uint(f);
  return (u & 0x80000000u) ? ~u : (u | 0x80000000u);
}
__device__ __forceinline__ float o2f(unsigned o) {
  return (o & 0x80000000u) ? __uint_as_float(o & 0x7FFFFFFFu)
                           : __uint_as_float(~o);
}

__global__ void k_init(unsigned* mm) {
  mm[0] = 0xFFFFFFFFu;  // ordered-min accumulator (== +inf)
  mm[1] = 0x00000000u;  // ordered-max accumulator (== -inf)
}

__global__ __launch_bounds__(256) void k_minmax(const float* __restrict__ x,
                                                long long n, unsigned* mm) {
  float lmin = INFINITY, lmax = -INFINITY;
  long long i = ((long long)blockIdx.x * blockDim.x + threadIdx.x) * 4;
  const long long stride = (long long)gridDim.x * blockDim.x * 4;
  for (; i < n; i += stride) {
    float4 v = *(const float4*)(x + i);
    lmin = fminf(lmin, fminf(fminf(v.x, v.y), fminf(v.z, v.w)));
    lmax = fmaxf(lmax, fmaxf(fmaxf(v.x, v.y), fmaxf(v.z, v.w)));
  }
  __shared__ unsigned smin[256], smax[256];
  const int tid = threadIdx.x;
  smin[tid] = f2o(lmin);
  smax[tid] = f2o(lmax);
  __syncthreads();
  for (int s = 128; s > 0; s >>= 1) {
    if (tid < s) {
      smin[tid] = min(smin[tid], smin[tid + s]);
      smax[tid] = max(smax[tid], smax[tid + s]);
    }
    __syncthreads();
  }
  if (tid == 0) {
    atomicMin(&mm[0], smin[0]);
    atomicMax(&mm[1], smax[0]);
  }
}

__global__ __launch_bounds__(256) void k_quant(const float* __restrict__ x,
                                               _Float16* __restrict__ qx,
                                               long long n,
                                               const unsigned* __restrict__ mm) {
  const float xmin = o2f(mm[0]);
  const float xmax = o2f(mm[1]);
  const float scale = (xmax - xmin) * (1.0f / 255.0f);
  const float inv = 1.0f / scale;
  // zp = clip(QMIN - round(xmin/scale), -128, 127); jnp.round == rintf
  const float zp = fminf(127.0f, fmaxf(-128.0f, -128.0f - rintf(xmin * inv)));
  long long i = ((long long)blockIdx.x * blockDim.x + threadIdx.x) * 4;
  const long long stride = (long long)gridDim.x * blockDim.x * 4;
  for (; i < n; i += stride) {
    float4 v = *(const float4*)(x + i);
    float q0 = fminf(127.0f, fmaxf(-128.0f, rintf(v.x * inv) + zp));
    float q1 = fminf(127.0f, fmaxf(-128.0f, rintf(v.y * inv) + zp));
    float q2 = fminf(127.0f, fmaxf(-128.0f, rintf(v.z * inv) + zp));
    float q3 = fminf(127.0f, fmaxf(-128.0f, rintf(v.w * inv) + zp));
    v4h h = {(_Float16)q0, (_Float16)q1, (_Float16)q2, (_Float16)q3};
    *(v4h*)(qx + i) = h;  // exact: integers in [-128,127]
  }
}

__global__ __launch_bounds__(256) void k_wsplit(const float* __restrict__ w,
                                                _Float16* __restrict__ whi,
                                                _Float16* __restrict__ wlo,
                                                long long n) {
  long long i = ((long long)blockIdx.x * blockDim.x + threadIdx.x) * 4;
  const long long stride = (long long)gridDim.x * blockDim.x * 4;
  for (; i < n; i += stride) {
    float4 v = *(const float4*)(w + i);
    _Float16 h0 = (_Float16)v.x, h1 = (_Float16)v.y;
    _Float16 h2 = (_Float16)v.z, h3 = (_Float16)v.w;
    v4h hi = {h0, h1, h2, h3};
    v4h lo = {(_Float16)(v.x - (float)h0), (_Float16)(v.y - (float)h1),
              (_Float16)(v.z - (float)h2), (_Float16)(v.w - (float)h3)};
    *(v4h*)(whi + i) = hi;
    *(v4h*)(wlo + i) = lo;
  }
}

// ---- WMMA GEMM -------------------------------------------------------------
// Block 256 threads = 8 waves. Block tile 128(M) x 128(N), BK = 32.
// Wave grid 2(M) x 4(N); each wave: 64x32 = 4x2 WMMA tiles of 16x16.
// LDS double-buffered; tiles filled with GLOBAL_LOAD_ASYNC_TO_LDS_B128.
#define BM 128
#define BN 128
#define BK 32
#define LDSW 40  // padded row stride in f16 (80B): conflict-free b128 ds loads

// Async 16-byte copy global -> LDS (per lane), tracked by ASYNCcnt.
__device__ __forceinline__ void async_cp16(unsigned lds_byte_addr,
                                           const void* gptr) {
  asm volatile("global_load_async_to_lds_b128 %0, %1, off"
               :
               : "v"(lds_byte_addr), "v"(gptr)
               : "memory");
}
__device__ __forceinline__ void wait_async0() {
  asm volatile("s_wait_asynccnt 0x0" ::: "memory");
}
__device__ __forceinline__ unsigned lds_addr(const void* p) {
  return (unsigned)(unsigned long long)p;  // AS3 offset = low 32 bits
}

__device__ __forceinline__ v16h ld_frag(const _Float16* p, int gap) {
  // 16 f16 as two b128 LDS loads; 'gap' = f16 distance between the halves.
  v8h lo = *(const v8h*)p;
  v8h hi = *(const v8h*)(p + gap);
  return __builtin_shufflevector(lo, hi, 0, 1, 2, 3, 4, 5, 6, 7, 8, 9, 10, 11,
                                 12, 13, 14, 15);
}

__device__ __forceinline__ void compute_step(
    const _Float16* __restrict__ pA, const _Float16* __restrict__ pBh,
    const _Float16* __restrict__ pBl, int waveM, int waveN, int lrow,
    int lhalf, v8f acc[4][2]) {
  // Issue ALL fragment loads first so s_wait_dscnt thresholds let later
  // loads complete underneath earlier WMMAs (no full-stall waits).
  // A frag (16x32 f16): lane holds M=lrow, K in {lhalf*8..+7} u {+16..+23}.
  // B frag (32x16): lane holds N=lrow, K in {lhalf*8..+7} u {+16..+23}; W is
  // (N,K) row-major so B "columns" are contiguous W rows: no transpose.
  v16h afrag[4], bfh[2], bfl[2];
#pragma unroll
  for (int mi = 0; mi < 4; ++mi)
    afrag[mi] =
        ld_frag(pA + (waveM * 64 + mi * 16 + lrow) * LDSW + lhalf * 8, 16);
#pragma unroll
  for (int ni = 0; ni < 2; ++ni) {
    bfh[ni] =
        ld_frag(pBh + (waveN * 32 + ni * 16 + lrow) * LDSW + lhalf * 8, 16);
    bfl[ni] =
        ld_frag(pBl + (waveN * 32 + ni * 16 + lrow) * LDSW + lhalf * 8, 16);
  }
  // 16 WMMAs; accumulator reuse distance is 8 (hi pass then lo pass).
#pragma unroll
  for (int ni = 0; ni < 2; ++ni)
#pragma unroll
    for (int mi = 0; mi < 4; ++mi)
      acc[mi][ni] = __builtin_amdgcn_wmma_f32_16x16x32_f16(
          false, afrag[mi], false, bfh[ni], (short)0, acc[mi][ni], false,
          false);
#pragma unroll
  for (int ni = 0; ni < 2; ++ni)
#pragma unroll
    for (int mi = 0; mi < 4; ++mi)
      acc[mi][ni] = __builtin_amdgcn_wmma_f32_16x16x32_f16(
          false, afrag[mi], false, bfl[ni], (short)0, acc[mi][ni], false,
          false);
}

__global__ __launch_bounds__(256) void k_gemm(
    const _Float16* __restrict__ qx, const _Float16* __restrict__ whi,
    const _Float16* __restrict__ wlo, const float* __restrict__ bias,
    float* __restrict__ out, int M, int N, int K) {
  __shared__ _Float16 sA[2][BM * LDSW];
  __shared__ _Float16 sBh[2][BN * LDSW];
  __shared__ _Float16 sBl[2][BN * LDSW];

  const int tid = threadIdx.x;
  const int lane = tid & 31;
  const int wave = tid >> 5;
  const int waveM = wave & 1;   // 0..1
  const int waveN = wave >> 1;  // 0..3
  const int blockM = blockIdx.y * BM;
  const int blockN = blockIdx.x * BN;

  const int lrow = lane & 15;   // M (A-frag) / N (B-frag) within 16x16 tile
  const int lhalf = lane >> 4;  // selects K-half per ISA fragment layout

  v8f acc[4][2];
#pragma unroll
  for (int mi = 0; mi < 4; ++mi)
#pragma unroll
    for (int ni = 0; ni < 2; ++ni)
#pragma unroll
      for (int r = 0; r < 8; ++r) acc[mi][ni][r] = 0.0f;

  // cooperative tile fill: thread t -> row t/2, 16-f16 chunk t%2 (32B/thread)
  const int cr = tid >> 1;
  const int cc = (tid & 1) * 16;
  const long long aRow = (long long)(blockM + cr) * K;
  const long long bRow = (long long)(blockN + cr) * K;
  const unsigned ldsOff = (unsigned)((cr * LDSW + cc) * 2);  // bytes
  const unsigned lA0 = lds_addr(sA[0]) + ldsOff;
  const unsigned lA1 = lds_addr(sA[1]) + ldsOff;
  const unsigned lH0 = lds_addr(sBh[0]) + ldsOff;
  const unsigned lH1 = lds_addr(sBh[1]) + ldsOff;
  const unsigned lL0 = lds_addr(sBl[0]) + ldsOff;
  const unsigned lL1 = lds_addr(sBl[1]) + ldsOff;

#define ISSUE_TILES(kk_, la_, lh_, ll_)                  \
  do {                                                   \
    const _Float16* ga = qx + aRow + (kk_) + cc;         \
    const _Float16* gh = whi + bRow + (kk_) + cc;        \
    const _Float16* gl = wlo + bRow + (kk_) + cc;        \
    async_cp16((la_), ga);                               \
    async_cp16((la_) + 16, ga + 8);                      \
    async_cp16((lh_), gh);                               \
    async_cp16((lh_) + 16, gh + 8);                      \
    async_cp16((ll_), gl);                               \
    async_cp16((ll_) + 16, gl + 8);                      \
  } while (0)

  // prologue: prefetch K-step 0 into buffer 0
  ISSUE_TILES(0, lA0, lH0, lL0);

  // Double-buffered main loop, unrolled x2 so the buffer index is static.
  // One barrier per K-step:
  //   wait_async (my buf-cur writes done) -> barrier (everyone's writes done
  //   AND everyone's reads of buf-other from last step done) -> prefetch into
  //   buf-other -> compute from buf-cur.
  for (int kk = 0; kk < K; kk += 2 * BK) {
    wait_async0();
    __syncthreads();
    ISSUE_TILES(kk + BK, lA1, lH1, lL1);  // kk+BK < K (K % 64 == 0)
    compute_step(sA[0], sBh[0], sBl[0], waveM, waveN, lrow, lhalf, acc);

    wait_async0();
    __syncthreads();
    if (kk + 2 * BK < K) ISSUE_TILES(kk + 2 * BK, lA0, lH0, lL0);
    compute_step(sA[1], sBh[1], sBl[1], waveM, waveN, lrow, lhalf, acc);
  }
#undef ISSUE_TILES

  // Epilogue: C/D layout -> lane col = lane%16, row = r + (lane/16)*8
#pragma unroll
  for (int ni = 0; ni < 2; ++ni) {
    const int col = blockN + waveN * 32 + ni * 16 + lrow;
    const float bv = bias[col];
#pragma unroll
    for (int mi = 0; mi < 4; ++mi) {
      const int row0 = blockM + waveM * 64 + mi * 16 + lhalf * 8;
#pragma unroll
      for (int r = 0; r < 8; ++r)
        out[(long long)(row0 + r) * N + col] = acc[mi][ni][r] + bv;
    }
  }
}

extern "C" void kernel_launch(void* const* d_in, const int* in_sizes, int n_in,
                              void* d_out, int out_size, void* d_ws,
                              size_t ws_size, hipStream_t stream) {
  const float* x = (const float*)d_in[0];     // (M, K) flattened
  const float* w = (const float*)d_in[1];     // (N, K)
  const float* bias = (const float*)d_in[2];  // (N)
  float* out = (float*)d_out;                 // (M, N)

  const long long N = in_sizes[2];
  const long long K = in_sizes[1] / N;
  const long long M = in_sizes[0] / K;
  const long long nx = M * K;
  const long long nw = N * K;

  // workspace: [mm: 2 u32][pad to 256][qx f16][whi f16][wlo f16] (~128 MiB)
  unsigned char* ws = (unsigned char*)d_ws;
  unsigned* mm = (unsigned*)ws;
  _Float16* qx = (_Float16*)(ws + 256);
  _Float16* whi = (_Float16*)(ws + 256 + nx * 2);
  _Float16* wlo = (_Float16*)(ws + 256 + nx * 2 + nw * 2);

  k_init<<<1, 1, 0, stream>>>(mm);
  k_minmax<<<1024, 256, 0, stream>>>(x, nx, mm);
  k_quant<<<2048, 256, 0, stream>>>(x, qx, nx, mm);
  k_wsplit<<<2048, 256, 0, stream>>>(w, whi, wlo, nw);

  dim3 grid((unsigned)(N / BN), (unsigned)(M / BM));
  k_gemm<<<grid, 256, 0, stream>>>(qx, whi, wlo, bias, out, (int)M, (int)N,
                                   (int)K);
}
